// GraphASTEncoder_10247791968326
// MI455X (gfx1250) — compile-verified, matrix operand order checked by reading, not run on hardware
//
#include <hip/hip_runtime.h>
#include <hip/hip_bf16.h>

#define H   128
#define H3  384
#define MSG_WAVES 4   // waves per message block
#define MSG_TPW   4   // edge tiles per wave (amortizes LDS weight staging)

// gfx1250 async global->LDS DMA (ASYNCcnt-tracked), probe-guarded so the build
// falls back to a VGPR-bounce copy if the toolchain lacks the builtins.
#if __has_builtin(__builtin_amdgcn_global_load_async_to_lds_b128) && \
    __has_builtin(__builtin_amdgcn_s_wait_asynccnt)
#define USE_ASYNC_LDS 1
#else
#define USE_ASYNC_LDS 0
#endif

typedef __attribute__((ext_vector_type(16))) __bf16 v16bf;
typedef __attribute__((ext_vector_type(8)))  __bf16 v8bf;
typedef __attribute__((ext_vector_type(8)))  float  v8f;
typedef __attribute__((ext_vector_type(4)))  float  v4f;
typedef __attribute__((ext_vector_type(2)))  int    v2i;

#if USE_ASYNC_LDS
// exact parameter types per hipcc diagnostic: GCC-vector int4, AS1 src / AS3 dst
typedef int b128_t __attribute__((vector_size(16)));
typedef __attribute__((address_space(1))) b128_t* gptr_b128;
typedef __attribute__((address_space(3))) b128_t* lptr_b128;
#endif

// ---------------------------------------------------------------------------
// WMMA helpers. 16-bit A 16x32 per-lane layout (ISA 7.12.2):
//   lanes 0-15  (kh=0): elems 0..7 -> K=kb*32+{0..7},  8..15 -> K=kb*32+{16..23}
//   lanes 16-31 (kh=1): elems 0..7 -> K=kb*32+{8..15}, 8..15 -> K=kb*32+{24..31}
// B (K x N) assumed symmetric: lane%16 selects column N, same K pattern.
// ---------------------------------------------------------------------------
static __device__ __forceinline__ v16bf pack_a_f32(const float* p, int kh) {
  v4f a0 = *(const v4f*)(p + kh * 8);
  v4f a1 = *(const v4f*)(p + kh * 8 + 4);
  v4f a2 = *(const v4f*)(p + 16 + kh * 8);
  v4f a3 = *(const v4f*)(p + 16 + kh * 8 + 4);
  v16bf r;
#pragma unroll
  for (int i = 0; i < 4; ++i) {
    r[i]      = (__bf16)a0[i];
    r[4 + i]  = (__bf16)a1[i];
    r[8 + i]  = (__bf16)a2[i];
    r[12 + i] = (__bf16)a3[i];
  }
  return r;
}

static __device__ __forceinline__ v16bf pack_a_bf16(const __bf16* p, int kh) {
  v8bf a0 = *(const v8bf*)(p + kh * 8);
  v8bf a1 = *(const v8bf*)(p + 16 + kh * 8);
  v16bf r;
#pragma unroll
  for (int i = 0; i < 8; ++i) { r[i] = a0[i]; r[8 + i] = a1[i]; }
  return r;
}

static __device__ __forceinline__ v8f wmma_bf16(v16bf a, v16bf b, v8f c) {
  return __builtin_amdgcn_wmma_f32_16x16x32_bf16(false, a, false, b, (short)0, c,
                                                 false, false);
}

// acc(16x16) = A(16xK) * W-rows  (gi[m,n] = sum_k x[m,k] * W[n,k])
template <int KB>
static __device__ __forceinline__ v8f mm_row(const __bf16* __restrict__ Bmat, int ldb,
                                             int nrow, const v16bf* A, int kh) {
  v8f acc = {0.f, 0.f, 0.f, 0.f, 0.f, 0.f, 0.f, 0.f};
  const __bf16* bp = Bmat + (size_t)nrow * ldb;
#pragma unroll
  for (int kb = 0; kb < KB; ++kb)
    acc = wmma_bf16(A[kb], pack_a_bf16(bp + kb * 32, kh), acc);
  return acc;
}

// ---------------------------------------------------------------------------
// Utility kernels
// ---------------------------------------------------------------------------
__global__ void k_zero(float* p, size_t n) {
  size_t i = (size_t)blockIdx.x * blockDim.x + threadIdx.x;
  if (i < n) p[i] = 0.f;
}

__global__ void k_f2bf(__bf16* d, const float* s, size_t n) {
  size_t i = (size_t)blockIdx.x * blockDim.x + threadIdx.x;
  if (i < n) d[i] = (__bf16)s[i];
}

__global__ void k_copy(float* d, const float* s, size_t n) {
  size_t i = (size_t)blockIdx.x * blockDim.x + threadIdx.x;
  if (i < n) d[i] = s[i];
}

__global__ void k_embed(float* h, __bf16* hbf, __bf16* res0, const float* emb,
                        const int* ids, int n_nodes) {
  size_t i = (size_t)blockIdx.x * blockDim.x + threadIdx.x;
  if (i < (size_t)n_nodes * H) {
    int n = (int)(i / H), j = (int)(i % H);
    float v = emb[(size_t)ids[n] * H + j];
    h[i] = v;
    __bf16 vb = (__bf16)v;
    hbf[i]  = vb;
    res0[i] = vb;
  }
}

__global__ void k_gather(float* d, const float* h, const int* idx, int cnt) {
  size_t i = (size_t)blockIdx.x * blockDim.x + threadIdx.x;
  if (i < (size_t)cnt * H) {
    int v = (int)(i / H), j = (int)(i % H);
    d[i] = h[(size_t)idx[v] * H + j];
  }
}

// ---------------------------------------------------------------------------
// Message kernel: grid-stride over 16-edge tiles, one tile per wave iteration:
//   incoming[dst] += h[src] @ W^T + b
// W (128x128 bf16) staged once per block in LDS (async DMA when available),
// reused for MSG_TPW tiles/wave. Full tiles take an unconditional-atomic path.
// ---------------------------------------------------------------------------
__global__ void __launch_bounds__(32 * MSG_WAVES)
k_msg(const __bf16* __restrict__ hbf, float* __restrict__ incoming,
      const int* __restrict__ edges, int n_edges, int src_off, int dst_off,
      const __bf16* __restrict__ Wbf, const float* __restrict__ bias) {
  __shared__ __bf16 Wl[H * H];
  __shared__ float  bl[H];
#if USE_ASYNC_LDS
  {
    // 32 KB in 16B chunks: cache->LDS DMA, no VGPR bounce, tracked by ASYNCcnt.
    const unsigned lbase = (unsigned)(size_t)(void*)&Wl[0];  // low 32b = LDS offset
    for (int c = threadIdx.x; c < (H * H * 2) / 16; c += blockDim.x) {
      __builtin_amdgcn_global_load_async_to_lds_b128(
          (gptr_b128)(size_t)((const char*)Wbf + (size_t)c * 16),
          (lptr_b128)(size_t)(lbase + (unsigned)c * 16),
          0, 0);
    }
    __builtin_amdgcn_s_wait_asynccnt(0);
  }
#else
  for (int i = threadIdx.x; i < (H * H) / 8; i += blockDim.x)
    ((v8bf*)Wl)[i] = ((const v8bf*)Wbf)[i];
#endif
  if ((int)threadIdx.x < H) bl[threadIdx.x] = bias[threadIdx.x];
  __syncthreads();

  const int wave = threadIdx.x >> 5;
  const int lane = threadIdx.x & 31;
  const int kh   = lane >> 4;
  const int m16  = lane & 15;
  const int tiles   = (n_edges + 15) >> 4;
  const int tstride = gridDim.x * MSG_WAVES;

  for (int tile = blockIdx.x * MSG_WAVES + wave; tile < tiles; tile += tstride) {
    const int  base = tile * 16;
    const bool full = (base + 16 <= n_edges);

    // prefetch next tile's edge pairs (gfx1250 global_prefetch_b8)
    if (tile + tstride < tiles)
      __builtin_prefetch(edges + (size_t)(tile + tstride) * 32 + lane, 0, 1);

    int ei = base + m16;
    if (ei >= n_edges) ei = n_edges - 1;
    const v2i ep  = ((const v2i*)edges)[ei];
    const int srow = src_off ? ep[1] : ep[0];
    const __bf16* hp = hbf + (size_t)srow * H;
    v16bf A[4];
#pragma unroll
    for (int kb = 0; kb < 4; ++kb) A[kb] = pack_a_bf16(hp + kb * 32, kh);

    int dstI[8];  // D-tile rows owned by this lane: M = r + 8*kh
#pragma unroll
    for (int r = 0; r < 8; ++r) {
      const int ee = base + r + 8 * kh;
      if (full || ee < n_edges) {
        const v2i ed = ((const v2i*)edges)[ee];
        dstI[r] = dst_off ? ed[1] : ed[0];
      } else {
        dstI[r] = -1;
      }
    }

    for (int nt = 0; nt < 8; ++nt) {
      v8f acc = {0.f, 0.f, 0.f, 0.f, 0.f, 0.f, 0.f, 0.f};
#pragma unroll
      for (int kb = 0; kb < 4; ++kb)
        acc = wmma_bf16(A[kb],
                        pack_a_bf16(&Wl[(size_t)(nt * 16 + m16) * H + kb * 32], kh),
                        acc);
      const int   col = nt * 16 + m16;
      const float bv  = bl[col];
      if (full) {  // wave-uniform: no per-lane exec juggling
#pragma unroll
        for (int r = 0; r < 8; ++r)
          atomicAdd(&incoming[(size_t)dstI[r] * H + col], acc[r] + bv);
      } else {
#pragma unroll
        for (int r = 0; r < 8; ++r)
          if (dstI[r] >= 0)
            atomicAdd(&incoming[(size_t)dstI[r] * H + col], acc[r] + bv);
      }
    }
  }
}

// ---------------------------------------------------------------------------
// GRU kernel: one wave per 16 nodes.  x = [incoming | residual(bf16)] (XDIM),
// gi = x@Wih^T, gh = hbf@Whh^T, gates, in-place update of h (f32) + hbf (bf16).
// ---------------------------------------------------------------------------
template <int XDIM>
__global__ void __launch_bounds__(256)
k_gru(float* __restrict__ h, __bf16* __restrict__ hbf,
      const float* __restrict__ incoming, const __bf16* __restrict__ resbf,
      const __bf16* __restrict__ Wih, const __bf16* __restrict__ Whh,
      const float* __restrict__ bih, const float* __restrict__ bhh, int n_nodes) {
  const int wave = threadIdx.x >> 5;
  const int lane = threadIdx.x & 31;
  const int kh   = lane >> 4;
  const int m16  = lane & 15;
  const int nodeBase = (blockIdx.x * 8 + wave) * 16;
  if (nodeBase >= n_nodes) return;
  const bool full = (nodeBase + 16 <= n_nodes);

  int row = nodeBase + m16;
  if (row >= n_nodes) row = n_nodes - 1;

  constexpr int KBX = XDIM / 32;
  v16bf Ax[KBX], Ah[4];
  const float* ip = incoming + (size_t)row * H;
#pragma unroll
  for (int kb = 0; kb < KBX; ++kb) {
    if (kb < 4) Ax[kb] = pack_a_f32(ip + kb * 32, kh);
    else        Ax[kb] = pack_a_bf16(resbf + (size_t)row * H + (kb - 4) * 32, kh);
  }
  const __bf16* hp = hbf + (size_t)row * H;
#pragma unroll
  for (int kb = 0; kb < 4; ++kb) Ah[kb] = pack_a_bf16(hp + kb * 32, kh);

  for (int nt = 0; nt < 8; ++nt) {
    const int nl = nt * 16 + m16;  // column within H
    v8f gir = mm_row<KBX>(Wih, XDIM, nl,         Ax, kh);
    v8f giz = mm_row<KBX>(Wih, XDIM, nl + H,     Ax, kh);
    v8f gin = mm_row<KBX>(Wih, XDIM, nl + 2 * H, Ax, kh);
    v8f ghr = mm_row<4>(Whh, H, nl,         Ah, kh);
    v8f ghz = mm_row<4>(Whh, H, nl + H,     Ah, kh);
    v8f ghn = mm_row<4>(Whh, H, nl + 2 * H, Ah, kh);
    const float bir = bih[nl], biz = bih[nl + H], bin = bih[nl + 2 * H];
    const float bhr = bhh[nl], bhz = bhh[nl + H], bhn = bhh[nl + 2 * H];
#pragma unroll
    for (int r = 0; r < 8; ++r) {
      const int node = nodeBase + r + 8 * kh;
      if (full || node < n_nodes) {
        const size_t off  = (size_t)node * H + nl;
        const float  hold = h[off];
        const float rr = 1.f / (1.f + __expf(-(gir[r] + bir + ghr[r] + bhr)));
        const float zz = 1.f / (1.f + __expf(-(giz[r] + biz + ghz[r] + bhz)));
        const float nn = tanhf(gin[r] + bin + rr * (ghn[r] + bhn));
        const float hv = (1.f - zz) * nn + zz * hold;
        h[off]   = hv;
        hbf[off] = (__bf16)hv;
      }
    }
  }
}

// ---------------------------------------------------------------------------
// Host orchestration
// ---------------------------------------------------------------------------
extern "C" void kernel_launch(void* const* d_in, const int* in_sizes, int n_in,
                              void* d_out, int out_size, void* d_ws, size_t ws_size,
                              hipStream_t stream) {
  const int N     = in_sizes[0];
  const int E_AST = in_sizes[1] / 2;
  const int E_M   = in_sizes[2] / 2;
  const int V     = in_sizes[3];
  const int P     = in_sizes[4];

  const int*   ids   = (const int*)d_in[0];
  const int*   eA    = (const int*)d_in[1];
  const int*   eM    = (const int*)d_in[2];
  const int*   varI  = (const int*)d_in[3];
  const int*   predI = (const int*)d_in[4];
  const float* emb   = (const float*)d_in[5];
  const float* msgW  = (const float*)d_in[6];
  const float* msgb  = (const float*)d_in[7];
  const float* WihA  = (const float*)d_in[8];
  const float* WihB  = (const float*)d_in[9];
  const float* WhhF  = (const float*)d_in[10];
  const float* bihF  = (const float*)d_in[11];
  const float* bhhF  = (const float*)d_in[12];
  float* out = (float*)d_out;

  const size_t nh = (size_t)N * H;
  float*  h      = (float*)d_ws;
  float*  inc    = h + nh;
  __bf16* hbf    = (__bf16*)(inc + nh);
  __bf16* res0   = hbf + nh;
  __bf16* res1   = res0 + nh;
  __bf16* msgWbf = res1 + nh;
  __bf16* Wihbf  = msgWbf + (size_t)4 * 4 * H * H;
  const size_t oA = (size_t)H3 * H, oB = (size_t)H3 * 2 * H;
  __bf16* Whhbf  = Wihbf + 2 * oA + 2 * oB;

  auto blocks = [](size_t n, size_t t) { return (unsigned)((n + t - 1) / t); };

  // weight conversions to bf16 (deterministic, cheap, done every call)
  k_f2bf<<<blocks((size_t)4 * 4 * H * H, 256), 256, 0, stream>>>(msgWbf, msgW, (size_t)4 * 4 * H * H);
  k_f2bf<<<blocks(oA, 256), 256, 0, stream>>>(Wihbf,               WihA,      oA);  // layer 0
  k_f2bf<<<blocks(oB, 256), 256, 0, stream>>>(Wihbf + oA,          WihB,      oB);  // layer 1
  k_f2bf<<<blocks(oA, 256), 256, 0, stream>>>(Wihbf + oA + oB,     WihA + oA, oA);  // layer 2
  k_f2bf<<<blocks(oB, 256), 256, 0, stream>>>(Wihbf + 2 * oA + oB, WihB + oB, oB);  // layer 3
  k_f2bf<<<blocks(4 * oA, 256), 256, 0, stream>>>(Whhbf, WhhF, 4 * oA);

  k_embed<<<blocks(nh, 256), 256, 0, stream>>>(h, hbf, res0, emb, ids, N);

  struct ET { const int* e; int n; int so; int dofs; };
  const ET et[4] = { {eA, E_AST, 0, 1}, {eA, E_AST, 1, 0},
                     {eM, E_M,   0, 1}, {eM, E_M,   1, 0} };
  const int LT[4] = {5, 2, 5, 2};
  const __bf16* WihL[4] = {Wihbf, Wihbf + oA, Wihbf + oA + oB, Wihbf + 2 * oA + oB};

  for (int l = 0; l < 4; ++l) {
    for (int t = 0; t < LT[l]; ++t) {
      k_zero<<<blocks(nh, 256), 256, 0, stream>>>(inc, nh);
      for (int e = 0; e < 4; ++e) {
        const int tiles = (et[e].n + 15) / 16;
        unsigned nb = blocks((size_t)tiles, MSG_WAVES * MSG_TPW);
        k_msg<<<nb, 32 * MSG_WAVES, 0, stream>>>(
            hbf, inc, et[e].e, et[e].n, et[e].so, et[e].dofs,
            msgWbf + (size_t)(l * 4 + e) * H * H, msgb + (size_t)(l * 4 + e) * H);
      }
      unsigned gb = blocks((size_t)N, 128);
      if (l == 0 || l == 2)
        k_gru<128><<<gb, 256, 0, stream>>>(h, hbf, inc, (const __bf16*)nullptr,
                                           WihL[l], Whhbf + (size_t)l * H3 * H,
                                           bihF + (size_t)l * H3, bhhF + (size_t)l * H3, N);
      else
        k_gru<256><<<gb, 256, 0, stream>>>(h, hbf, inc, (l == 1) ? res0 : res1,
                                           WihL[l], Whhbf + (size_t)l * H3 * H,
                                           bihF + (size_t)l * H3, bhhF + (size_t)l * H3, N);
    }
    if (l == 0)  // node_states[1] = h after layer 0 (residual for layer 3)
      k_f2bf<<<blocks(nh, 256), 256, 0, stream>>>(res1, h, nh);
  }

  k_copy<<<blocks(nh, 256), 256, 0, stream>>>(out, h, nh);
  k_gather<<<blocks((size_t)V * H, 256), 256, 0, stream>>>(out + nh, h, varI, V);
  k_gather<<<blocks((size_t)P * H, 256), 256, 0, stream>>>(out + nh + (size_t)V * H, h, predI, P);
}
